// CrossAttention_44951127719995
// MI455X (gfx1250) — matrix operand order repaired and checked by reading, not compiled, not run on hardware
//
#include <hip/hip_runtime.h>

typedef __attribute__((ext_vector_type(16))) _Float16 v16h;
typedef __attribute__((ext_vector_type(8)))  _Float16 v8h;
typedef __attribute__((ext_vector_type(2)))  _Float16 v2h;
typedef __attribute__((ext_vector_type(8)))  float    v8f;

// ---------------------------------------------------------------------------
// Fragment-ordered LDS layout (matches CDNA5 WMMA VGPR layouts, wave32):
//   A 16x32 (MxK): lane=(m&15)|((k>>3&1)<<4), e=(k&7)|((k>>4)<<3)
//   B 32x16 (KxN): lane=(n&15)|((k>>3&1)<<4), e=(k&7)|((k>>4)<<3)
// 8 consecutive k (k%8==0) are contiguous in BOTH global memory and the LDS
// fragment -> one 16B async copy moves a whole chunk.
// ---------------------------------------------------------------------------

__device__ __forceinline__ int a_idx(int m, int k) {
  int t    = m >> 4;
  int lane = (m & 15) | (((k >> 3) & 1) << 4);
  int e    = (k & 7) | ((k >> 4) << 3);
  return t * 512 + lane * 16 + e;
}

__device__ __forceinline__ int b_idx(int k, int n) {
  int t    = n >> 4;
  int lane = (n & 15) | (((k >> 3) & 1) << 4);
  return t * 512 + lane * 16 + ((k & 7) | ((k >> 4) << 3));
}

__device__ __forceinline__ v16h frag_ld(const _Float16* base, int tile) {
  int lane = threadIdx.x & 31;
  return *(const v16h*)(base + tile * 512 + lane * 16);
}

__device__ __forceinline__ v8f wmma_step(v16h a, v16h b, v8f c) {
  return __builtin_amdgcn_wmma_f32_16x16x32_f16(false, a, false, b, (short)0, c, false, false);
}

// 16B global -> LDS async copy (ASYNCcnt-tracked, no VGPR staging)
__device__ __forceinline__ void async_b128(unsigned lds_off, const _Float16* g) {
  asm volatile("global_load_async_to_lds_b128 %0, %1, off"
               :: "v"(lds_off), "v"(g) : "memory");
}

template <int N>
__device__ __forceinline__ void wait_async() {
#if __has_builtin(__builtin_amdgcn_s_wait_asynccnt)
  __builtin_amdgcn_s_wait_asynccnt(N);
#else
  asm volatile("s_wait_asynccnt %0" :: "i"(N) : "memory");
#endif
}

// ---- proj: block tile 128x64, double-buffered (12KB x2) ----
#define P_BUF_ELEMS 6144
#define P_B_REGION  4096

// ---- qk/pv: block tile 256x64, 3-deep async pipeline (20KB x3 = 60KB) ----
#define T_BUF_ELEMS 10240
#define T_BUF_BYTES 20480
#define T_B_REGION  8192     // f16 offset of B region
#define T_B_RBYTES  16384

// ---- Kernel 1: C_f16[M,N] = A_f32[M,K] @ W_f32[K,N] + bias (fp32 -> f16) ----
__global__ __launch_bounds__(256) void proj_kernel(const float* __restrict__ A,
                                                   const float* __restrict__ W,
                                                   const float* __restrict__ bias,
                                                   _Float16* __restrict__ C,
                                                   int M, int N, int K) {
  __shared__ v16h buf[2][384];
  _Float16* lds = (_Float16*)&buf[0][0];
  int tid = threadIdx.x;
  int gm0 = blockIdx.y * 128, gn0 = blockIdx.x * 64;
  int wave = tid >> 5;
  (void)M;

  int aIdx[8]; const float* agp[8];
#pragma unroll
  for (int i = 0; i < 8; ++i) {
    int pidx = tid + i * 256;
    int m = pidx >> 4, k = (pidx & 15) * 2;
    aIdx[i] = a_idx(m, k);
    agp[i]  = A + (size_t)(gm0 + m) * K + k;
  }
  int bIdx[4]; const float* bgp[4];
#pragma unroll
  for (int i = 0; i < 4; ++i) {
    int pidx = tid + i * 256;
    int n = pidx & 63, k = (pidx >> 6) * 2;
    bIdx[i] = P_B_REGION + b_idx(k, n);
    bgp[i]  = W + (size_t)k * N + (gn0 + n);
  }

#pragma unroll
  for (int i = 0; i < 8; ++i) {
    const float* ap = agp[i];
    *(v2h*)(lds + aIdx[i]) = (v2h){(_Float16)ap[0], (_Float16)ap[1]};
  }
#pragma unroll
  for (int i = 0; i < 4; ++i) {
    const float* wp = bgp[i];
    *(v2h*)(lds + bIdx[i]) = (v2h){(_Float16)wp[0], (_Float16)wp[N]};
  }
  __syncthreads();

  v8f acc[4] = {};
  const int NS = K / 32;
  for (int s = 0; s < NS; ++s) {
    int cur = s & 1, nxt = cur ^ 1;
    v2h aN[8], bN[4];
    if (s + 1 < NS) {
      int ka = (s + 1) * 32;
#pragma unroll
      for (int i = 0; i < 8; ++i) {
        const float* ap = agp[i] + ka;
        aN[i] = (v2h){(_Float16)ap[0], (_Float16)ap[1]};
      }
#pragma unroll
      for (int i = 0; i < 4; ++i) {
        const float* wp = bgp[i] + (size_t)ka * N;
        bN[i] = (v2h){(_Float16)wp[0], (_Float16)wp[N]};
      }
    }
    const _Float16* Ab = lds + (size_t)cur * P_BUF_ELEMS;
    const _Float16* Bb = Ab + P_B_REGION;
    v16h a = frag_ld(Ab, wave);
    v16h bf[4];
#pragma unroll
    for (int t = 0; t < 4; ++t) bf[t] = frag_ld(Bb, t);
#pragma unroll
    for (int t = 0; t < 4; ++t) acc[t] = wmma_step(a, bf[t], acc[t]);
    if (s + 1 < NS) {
      _Float16* dst = lds + (size_t)nxt * P_BUF_ELEMS;
#pragma unroll
      for (int i = 0; i < 8; ++i) *(v2h*)(dst + aIdx[i]) = aN[i];
#pragma unroll
      for (int i = 0; i < 4; ++i) *(v2h*)(dst + bIdx[i]) = bN[i];
    }
    __syncthreads();
  }

  int lane = tid & 31, half = lane >> 4, nn = lane & 15;
#pragma unroll
  for (int t = 0; t < 4; ++t) {
    int c = gn0 + t * 16 + nn;
    float bv = bias[c];
#pragma unroll
    for (int r = 0; r < 8; ++r) {
      int m = gm0 + wave * 16 + half * 8 + r;
      C[(size_t)m * N + c] = (_Float16)(acc[t][r] + bv);
    }
  }
}

// ---- Kernel 2: S_f16 = scale * Q @ K^T (d=768), 256x64 tile, async x3 ----
// Batch = 5 async_b128/thread (4 A + 1 B); in-order completion =>
// end of step s: wait asynccnt<=5 -> batch s+1 resident, s+2 in flight.
__global__ __launch_bounds__(256) void qk_kernel(const _Float16* __restrict__ Q,
                                                 const _Float16* __restrict__ Kf,
                                                 _Float16* __restrict__ S,
                                                 int Sq, int Sk, float scale) {
  __shared__ v16h buf[3][640];
  _Float16* lds = (_Float16*)&buf[0][0];
  unsigned lbase = (unsigned)(size_t)(void*)lds;
  int tid = threadIdx.x;
  const _Float16* Qb = Q  + (size_t)blockIdx.z * Sq * 768;
  const _Float16* Kb = Kf + (size_t)blockIdx.z * Sk * 768;
  _Float16*       Sb = S  + (size_t)blockIdx.z * Sq * Sk;
  int gm0 = blockIdx.y * 256, gn0 = blockIdx.x * 64;
  int wave = tid >> 5;

  // A tile 256x32 = 1024 x 16B chunks (4/thread); B tile = 256 chunks (1/thread)
  unsigned aoff[4]; const _Float16* agp[4];
#pragma unroll
  for (int i = 0; i < 4; ++i) {
    int c = tid + i * 256;
    int m = c >> 2, kc = (c & 3) * 8;
    aoff[i] = (unsigned)(a_idx(m, kc) * 2);
    agp[i]  = Qb + (size_t)(gm0 + m) * 768 + kc;
  }
  int nB = tid >> 2, kcB = (tid & 3) * 8;
  unsigned boff = (unsigned)(T_B_RBYTES + b_idx(kcB, nB) * 2);
  const _Float16* bgp = Kb + (size_t)(gn0 + nB) * 768 + kcB;

  // prologue: async-fill buffers 0 and 1
#pragma unroll
  for (int s0 = 0; s0 < 2; ++s0) {
    int ka = s0 * 32;
    unsigned bo = (unsigned)s0 * T_BUF_BYTES;
#pragma unroll
    for (int i = 0; i < 4; ++i) async_b128(lbase + bo + aoff[i], agp[i] + ka);
    async_b128(lbase + bo + boff, bgp + ka);
  }
  wait_async<5>();
  __syncthreads();

  v8f acc0[4] = {}, acc1[4] = {};
  const int NS = 768 / 32;
  for (int s = 0; s < NS; ++s) {
    int cur = s % 3;
    const _Float16* Ab = lds + (size_t)cur * T_BUF_ELEMS;
    const _Float16* Bb = Ab + T_B_REGION;
    v16h a0 = frag_ld(Ab, 2 * wave);
    v16h a1 = frag_ld(Ab, 2 * wave + 1);
    v16h bf[4];
#pragma unroll
    for (int t = 0; t < 4; ++t) bf[t] = frag_ld(Bb, t);
#pragma unroll
    for (int t = 0; t < 4; ++t) {
      acc0[t] = wmma_step(a0, bf[t], acc0[t]);
      acc1[t] = wmma_step(a1, bf[t], acc1[t]);
    }
    if (s + 2 < NS) {
      int nb = (s + 2) % 3, ka = (s + 2) * 32;
      unsigned bo = (unsigned)nb * T_BUF_BYTES;
#pragma unroll
      for (int i = 0; i < 4; ++i) async_b128(lbase + bo + aoff[i], agp[i] + ka);
      async_b128(lbase + bo + boff, bgp + ka);
      wait_async<5>();
    } else {
      wait_async<0>();
    }
    __syncthreads();
  }

  int lane = tid & 31, half = lane >> 4, nn = lane & 15;
#pragma unroll
  for (int t = 0; t < 4; ++t) {
    int c = gn0 + t * 16 + nn;
#pragma unroll
    for (int r = 0; r < 8; ++r) {
      int m0 = gm0 + (2 * wave) * 16 + half * 8 + r;
      Sb[(size_t)m0 * Sk + c]        = (_Float16)(acc0[t][r] * scale);
      Sb[(size_t)(m0 + 16) * Sk + c] = (_Float16)(acc1[t][r] * scale);
    }
  }
}

// ---- Kernel 3: in-place row softmax, vectorized v8h, one read + one write ----
__global__ __launch_bounds__(256) void softmax_kernel(_Float16* __restrict__ S, int Sk) {
  __shared__ float red[256];
  int tid = threadIdx.x;
  _Float16* row = S + (size_t)blockIdx.x * Sk;
  int cnt = Sk >> 11;                      // 2048 -> 1, 4096 -> 2 vectors/thread
  v8h xv[2];
  float m = -1e30f;
  for (int i = 0; i < cnt; ++i) {
    xv[i] = ((const v8h*)row)[tid + (i << 8)];
#pragma unroll
    for (int j = 0; j < 8; ++j) m = fmaxf(m, (float)xv[i][j]);
  }
  red[tid] = m; __syncthreads();
  for (int s = 128; s > 0; s >>= 1) { if (tid < s) red[tid] = fmaxf(red[tid], red[tid + s]); __syncthreads(); }
  float rmax = red[0]; __syncthreads();
  float fv[16];
  float sum = 0.f;
  for (int i = 0; i < cnt; ++i) {
#pragma unroll
    for (int j = 0; j < 8; ++j) {
      float e = __expf((float)xv[i][j] - rmax);
      fv[i * 8 + j] = e;
      sum += e;
    }
  }
  red[tid] = sum; __syncthreads();
  for (int s = 128; s > 0; s >>= 1) { if (tid < s) red[tid] += red[tid + s]; __syncthreads(); }
  float inv = 1.0f / red[0];
  for (int i = 0; i < cnt; ++i) {
    v8h o;
#pragma unroll
    for (int j = 0; j < 8; ++j) o[j] = (_Float16)(fv[i * 8 + j] * inv);
    ((v8h*)row)[tid + (i << 8)] = o;
  }
}

// ---- Kernel 4: O_f32 = P @ V, 256x64 tile; P async (4/batch), V regular ----
__global__ __launch_bounds__(256) void pv_kernel(const _Float16* __restrict__ P,
                                                 const _Float16* __restrict__ V,
                                                 float* __restrict__ O,
                                                 int Sq, int Sk) {
  __shared__ v16h buf[3][640];
  _Float16* lds = (_Float16*)&buf[0][0];
  unsigned lbase = (unsigned)(size_t)(void*)lds;
  int tid = threadIdx.x;
  const _Float16* Pb = P + (size_t)blockIdx.z * Sq * Sk;
  const _Float16* Vb = V + (size_t)blockIdx.z * Sk * 768;
  float*          Ob = O + (size_t)blockIdx.z * Sq * 768;
  int gm0 = blockIdx.y * 256, gn0 = blockIdx.x * 64;
  int wave = tid >> 5;

  unsigned aoff[4]; const _Float16* agp[4];
#pragma unroll
  for (int i = 0; i < 4; ++i) {
    int c = tid + i * 256;
    int m = c >> 2, kc = (c & 3) * 8;
    aoff[i] = (unsigned)(a_idx(m, kc) * 2);
    agp[i]  = Pb + (size_t)(gm0 + m) * Sk + kc;
  }
  int bIdx[4]; const _Float16* bgp[4];
#pragma unroll
  for (int i = 0; i < 4; ++i) {
    int pidx = tid + i * 256;
    int n = pidx & 63, k = (pidx >> 6) * 2;
    bIdx[i] = T_B_REGION + b_idx(k, n);
    bgp[i]  = Vb + (size_t)k * 768 + (gn0 + n);
  }

  // prologue: batches 0 and 1
#pragma unroll
  for (int s0 = 0; s0 < 2; ++s0) {
    int ka = s0 * 32;
    unsigned bo = (unsigned)s0 * T_BUF_BYTES;
#pragma unroll
    for (int i = 0; i < 4; ++i) async_b128(lbase + bo + aoff[i], agp[i] + ka);
    size_t kadv = (size_t)ka * 768;
    _Float16* dst = lds + (size_t)s0 * T_BUF_ELEMS;
#pragma unroll
    for (int i = 0; i < 4; ++i) {
      const _Float16* vp = bgp[i] + kadv;
      *(v2h*)(dst + bIdx[i]) = (v2h){vp[0], vp[768]};
    }
  }
  wait_async<4>();
  __syncthreads();

  v8f acc0[4] = {}, acc1[4] = {};
  const int NS = Sk / 32;
  for (int s = 0; s < NS; ++s) {
    int cur = s % 3;
    v2h bN[4];
    bool pf = (s + 2 < NS);
    if (pf) {
      size_t kadv = (size_t)(s + 2) * 32 * 768;
#pragma unroll
      for (int i = 0; i < 4; ++i) {
        const _Float16* vp = bgp[i] + kadv;
        bN[i] = (v2h){vp[0], vp[768]};
      }
    }
    const _Float16* Ab = lds + (size_t)cur * T_BUF_ELEMS;
    const _Float16* Bb = Ab + T_B_REGION;
    v16h a0 = frag_ld(Ab, 2 * wave);
    v16h a1 = frag_ld(Ab, 2 * wave + 1);
    v16h bf[4];
#pragma unroll
    for (int t = 0; t < 4; ++t) bf[t] = frag_ld(Bb, t);
#pragma unroll
    for (int t = 0; t < 4; ++t) {
      acc0[t] = wmma_step(a0, bf[t], acc0[t]);
      acc1[t] = wmma_step(a1, bf[t], acc1[t]);
    }
    if (pf) {
      int nb = (s + 2) % 3, ka = (s + 2) * 32;
      unsigned bo = (unsigned)nb * T_BUF_BYTES;
#pragma unroll
      for (int i = 0; i < 4; ++i) async_b128(lbase + bo + aoff[i], agp[i] + ka);
      _Float16* dst = lds + (size_t)nb * T_BUF_ELEMS;
#pragma unroll
      for (int i = 0; i < 4; ++i) *(v2h*)(dst + bIdx[i]) = bN[i];
      wait_async<4>();
    } else {
      wait_async<0>();
    }
    __syncthreads();
  }

  int lane = tid & 31, half = lane >> 4, nn = lane & 15;
#pragma unroll
  for (int t = 0; t < 4; ++t) {
    int c = gn0 + t * 16 + nn;
#pragma unroll
    for (int r = 0; r < 8; ++r) {
      int m0 = gm0 + (2 * wave) * 16 + half * 8 + r;
      Ob[(size_t)m0 * 768 + c]        = acc0[t][r];
      Ob[(size_t)(m0 + 16) * 768 + c] = acc1[t][r];
    }
  }
}

extern "C" void kernel_launch(void* const* d_in, const int* in_sizes, int n_in,
                              void* d_out, int out_size, void* d_ws, size_t ws_size,
                              hipStream_t stream) {
  (void)in_sizes; (void)n_in; (void)out_size; (void)ws_size;
  const float* syn = (const float*)d_in[0];   // [4,4096,768]
  const float* sem = (const float*)d_in[1];   // [4,2048,768]
  const float* Wq1 = (const float*)d_in[2];  const float* bq1 = (const float*)d_in[3];
  const float* Wk1 = (const float*)d_in[4];  const float* bk1 = (const float*)d_in[5];
  const float* Wv1 = (const float*)d_in[6];  const float* bv1 = (const float*)d_in[7];
  const float* Wq2 = (const float*)d_in[8];  const float* bq2 = (const float*)d_in[9];
  const float* Wk2 = (const float*)d_in[10]; const float* bk2 = (const float*)d_in[11];
  const float* Wv2 = (const float*)d_in[12]; const float* bv2 = (const float*)d_in[13];

  const int B = 4, Ssyn = 4096, Ssem = 2048, H = 768;
  _Float16* ws = (_Float16*)d_ws;
  size_t o = 0;
  _Float16* Q1 = ws + o; o += (size_t)B * Ssem * H;
  _Float16* K1 = ws + o; o += (size_t)B * Ssyn * H;
  _Float16* V1 = ws + o; o += (size_t)B * Ssyn * H;
  _Float16* Q2 = ws + o; o += (size_t)B * Ssyn * H;
  _Float16* K2 = ws + o; o += (size_t)B * Ssem * H;
  _Float16* V2 = ws + o; o += (size_t)B * Ssem * H;
  _Float16* S1 = ws + o; o += (size_t)B * Ssem * Ssyn;
  _Float16* S2 = ws + o; o += (size_t)B * Ssyn * Ssem;

  float* O1 = (float*)d_out;                       // enhanced_semantic [4,2048,768]
  float* O2 = O1 + (size_t)B * Ssem * H;           // enhanced_syntax  [4,4096,768]

  const float scale = 0.036084391824351615f;       // 768^-0.5
  dim3 blk(256);

  proj_kernel<<<dim3(H / 64, (B * Ssem) / 128), blk, 0, stream>>>(sem, Wq1, bq1, Q1, B * Ssem, H, H);
  proj_kernel<<<dim3(H / 64, (B * Ssyn) / 128), blk, 0, stream>>>(syn, Wk1, bk1, K1, B * Ssyn, H, H);
  proj_kernel<<<dim3(H / 64, (B * Ssyn) / 128), blk, 0, stream>>>(syn, Wv1, bv1, V1, B * Ssyn, H, H);
  proj_kernel<<<dim3(H / 64, (B * Ssyn) / 128), blk, 0, stream>>>(syn, Wq2, bq2, Q2, B * Ssyn, H, H);
  proj_kernel<<<dim3(H / 64, (B * Ssem) / 128), blk, 0, stream>>>(sem, Wk2, bk2, K2, B * Ssem, H, H);
  proj_kernel<<<dim3(H / 64, (B * Ssem) / 128), blk, 0, stream>>>(sem, Wv2, bv2, V2, B * Ssem, H, H);

  qk_kernel<<<dim3(Ssyn / 64, Ssem / 256, B), blk, 0, stream>>>(Q1, K1, S1, Ssem, Ssyn, scale);
  softmax_kernel<<<dim3(B * Ssem), blk, 0, stream>>>(S1, Ssyn);
  pv_kernel<<<dim3(H / 64, Ssem / 256, B), blk, 0, stream>>>(S1, V1, O1, Ssem, Ssyn);

  qk_kernel<<<dim3(Ssem / 64, Ssyn / 256, B), blk, 0, stream>>>(Q2, K2, S2, Ssyn, Ssem, scale);
  softmax_kernel<<<dim3(B * Ssyn), blk, 0, stream>>>(S2, Ssem);
  pv_kernel<<<dim3(H / 64, Ssyn / 256, B), blk, 0, stream>>>(S2, V2, O2, Ssyn, Ssem);
}
